// SymmetricHierarchicalLayer_6828998000912
// MI455X (gfx1250) — compile-verified, matrix operand order checked by reading, not run on hardware
//
#include <hip/hip_runtime.h>

// ---------------------------------------------------------------------------
// SymmetricHierarchicalLayer for MI455X (gfx1250, wave32, WMMA bf16)
// Fragment-major weight swizzle: every WMMA operand load is a contiguous
// 32B/lane vector load (no scalar u16 gathers, no packing moves).
// ---------------------------------------------------------------------------
constexpr int N_ = 50000, F_ = 64, R_ = 16, B_ = 5, H_ = 128;
constexpr int E_ = 800000, K_ = 25000;
constexpr int KPAD_ = (K_ + 15) & ~15;  // 25008: pad pooled rows so WMMA tiles are unguarded

#define DEV __device__ __forceinline__

typedef __attribute__((ext_vector_type(16))) __bf16 v16bf;
typedef __attribute__((ext_vector_type(8)))  float  v8f;

DEV v8f wmma_bf16(v16bf a, v16bf b, v8f c) {
  // 8 args: (neg_a, A, neg_b, B, c_mod, C, reuse_a, reuse_b)
  return __builtin_amdgcn_wmma_f32_16x16x32_bf16(false, a, false, b, (short)0, c,
                                                 false, false);
}

DEV void atomAddF(float* p, float v) {
  __hip_atomic_fetch_add(p, v, __ATOMIC_RELAXED, __HIP_MEMORY_SCOPE_AGENT);
}

// Fragment-major swizzle. ISA 16-bit A layout: lane holds row (lane&15),
// K-pair block ko=(lane>>4)*8; vgpr v covers k = kb*32 + ((v>>2)<<4) + ko + 2*(v&3) {+1}.
// B uses the identical mapping with rc = column. One 16x32 (or 32x16) fragment
// = 32 lanes x 16 bf16, stored lane-contiguous: frag*512 + lane*16 + elem.
DEV int frag_idx(int k, int rc) {
  int kk = k & 31;
  int lane = (rc & 15) + (((kk >> 3) & 1) << 4);
  int v = ((kk >> 4) << 2) | ((kk & 7) >> 1);
  return (k >> 5) * 512 + lane * 16 + 2 * v + (k & 1);
}

// ---------------------------------------------------------------------------
// Param prep: W_r = sum_b comp[r,b]*bases[b] (r==nrel -> root), f32 -> bf16,
// written directly in fragment-major order: [r][nt][kb][lane][16].
// ---------------------------------------------------------------------------
__global__ void k_prep_W_swz(const float* __restrict__ bases, const float* __restrict__ comp,
                             const float* __restrict__ root, __bf16* __restrict__ Wsw,
                             int nrel) {
  int r = blockIdx.x;
  for (int i = threadIdx.x; i < F_ * F_; i += blockDim.x) {
    int k = i >> 6, n = i & 63;
    float v;
    if (r == nrel) {
      v = root[i];
    } else {
      v = 0.f;
#pragma unroll
      for (int b = 0; b < B_; ++b) v += comp[r * B_ + b] * bases[b * F_ * F_ + i];
    }
    // K=64 -> NKb=2 -> nt stride = 2*512
    Wsw[r * F_ * F_ + (n >> 4) * 1024 + frag_idx(k, n)] = (__bf16)v;
  }
}

// row-major [K][Ncols] f32 -> fragment-major bf16
__global__ void k_f2bf_swz(const float* __restrict__ in, __bf16* __restrict__ out,
                           int K, int Ncols) {
  int i = blockIdx.x * 256 + threadIdx.x;
  if (i >= K * Ncols) return;
  int k = i / Ncols, n = i % Ncols;
  int NKb = K >> 5;
  out[(n >> 4) * (NKb * 512) + frag_idx(k, n)] = (__bf16)in[i];
}

__global__ void k_zero_f(float* p, int n) {
  int i = blockIdx.x * 256 + threadIdx.x;
  if (i < n) p[i] = 0.f;
}
__global__ void k_zero_i(int* p, int n) {
  int i = blockIdx.x * 256 + threadIdx.x;
  if (i < n) p[i] = 0;
}

// ---------------------------------------------------------------------------
// xw[n, r, :] = x[n, :] @ W_r for r in [0, nmat). One 16-row tile per block.
// Rows must be padded to a multiple of 16 (all accesses unguarded).
// A staged in LDS in fragment layout; B loads are contiguous 32B/lane.
// ---------------------------------------------------------------------------
__global__ void __launch_bounds__(256) k_xw(const float* __restrict__ x,
                                            const __bf16* __restrict__ Wsw,
                                            float* __restrict__ xw, int nmat) {
  __shared__ __bf16 Af[2 * 512];  // 16x64 tile as 2 K-block fragments
  int row0 = blockIdx.x * 16;
  int t = threadIdx.x;
  for (int i = t; i < 16 * 64; i += 256) {
    int m = i >> 6, k = i & 63;
    Af[frag_idx(k, m)] = (__bf16)x[(row0 + m) * 64 + k];
  }
  __syncthreads();
  int lane = t & 31, wid = t >> 5;
  v16bf a0 = *(const v16bf*)&Af[lane * 16];
  v16bf a1 = *(const v16bf*)&Af[512 + lane * 16];
  for (int r = wid; r < nmat; r += 8) {
    const __bf16* W = Wsw + r * F_ * F_;
#pragma unroll
    for (int nt = 0; nt < 4; ++nt) {
      v8f acc = {};
      acc = wmma_bf16(a0, *(const v16bf*)(W + nt * 1024 + lane * 16), acc);
      acc = wmma_bf16(a1, *(const v16bf*)(W + nt * 1024 + 512 + lane * 16), acc);
      int col = nt * 16 + (lane & 15);
      int rbase = row0 + ((lane >> 4) << 3);
#pragma unroll
      for (int j = 0; j < 8; ++j)
        xw[((rbase + j) * nmat + r) * 64 + col] = acc[j];
    }
  }
}

// ---------------------------------------------------------------------------
// R-GCN edge passes (per-(dst,rel) mean folded into scatter weight)
// ---------------------------------------------------------------------------
__global__ void k_edge_cnt1(const int* __restrict__ src, const int* __restrict__ dst,
                            const int* __restrict__ et, float* __restrict__ cnt, int E) {
  int e = blockIdx.x * 256 + threadIdx.x;
  if (e < E) atomAddF(&cnt[dst[e] * R_ + et[e]], 1.0f);
}

__global__ void k_init_h(const float* __restrict__ xw, const float* __restrict__ bias,
                         float* __restrict__ h, int Nn, int nmat) {
  int i = blockIdx.x * 256 + threadIdx.x;
  if (i < Nn * 64) {
    int n = i >> 6, f = i & 63;
    h[i] = xw[(n * nmat + (nmat - 1)) * 64 + f] + bias[f];  // root term + bias
  }
}

__global__ void k_scatter1(const int* __restrict__ src, const int* __restrict__ dst,
                           const int* __restrict__ et, const float* __restrict__ conf,
                           const float* __restrict__ xw, const float* __restrict__ cnt,
                           float* __restrict__ h, int E) {
  int idx = blockIdx.x * 256 + threadIdx.x;
  int e = idx >> 6, f = idx & 63;
  if (e >= E) return;
  int d = dst[e], s = src[e], r = et[e];
  float w = conf[e] / fmaxf(cnt[d * R_ + r], 1.0f);
  atomAddF(&h[d * 64 + f], xw[(s * 17 + r) * 64 + f] * w);
}

// ---------------------------------------------------------------------------
// LayerNorm(64) + ReLU : one wave32 per row, 2 elems/lane, shuffle reduce
// ---------------------------------------------------------------------------
__global__ void k_ln_relu(const float* __restrict__ in, const float* __restrict__ g,
                          const float* __restrict__ b, float* __restrict__ out, int Nn) {
  int t = threadIdx.x, lane = t & 31, wid = t >> 5;
  int row = blockIdx.x * 8 + wid;
  if (row >= Nn) return;
  float v0 = in[row * 64 + lane], v1 = in[row * 64 + 32 + lane];
  float s = v0 + v1, s2 = v0 * v0 + v1 * v1;
#pragma unroll
  for (int m = 16; m; m >>= 1) {
    s += __shfl_xor(s, m, 32);
    s2 += __shfl_xor(s2, m, 32);
  }
  float mean = s * (1.f / 64.f);
  float var = s2 * (1.f / 64.f) - mean * mean;
  float inv = rsqrtf(var + 1e-5f);
  out[row * 64 + lane]      = fmaxf((v0 - mean) * inv * g[lane] + b[lane], 0.f);
  out[row * 64 + 32 + lane] = fmaxf((v1 - mean) * inv * g[32 + lane] + b[32 + lane], 0.f);
}

// ---------------------------------------------------------------------------
// Coupling layer (forward & inverse), in place. One wave per 16-row tile.
//   p = passive half, a = active half.  h = relu(p@w1+b1); [s|t] = h@w2+b2
//   fwd: a = a*exp(tanh(s)) + t      inv: a = (a - t)*exp(-tanh(s))
// Nn must be a multiple of 16. Weights pre-swizzled fragment-major:
//   w1sw: [nt(8)][kb(1)][512] ; w2sw: [nt(4)][kb(4)][512]
// Per-wave LDS regions; same-wave DS ops are in-order -> no barrier needed.
// ---------------------------------------------------------------------------
__global__ void __launch_bounds__(128) k_coupling(float* __restrict__ x,
                                                  const __bf16* __restrict__ w1sw,
                                                  const float* __restrict__ b1,
                                                  const __bf16* __restrict__ w2sw,
                                                  const float* __restrict__ b2,
                                                  int parity, int inverse, int Nn) {
  __shared__ __bf16 Pf[4][512];   // passive 16x32 tile, fragment layout
  __shared__ __bf16 Hf[4][2048];  // h 16x128 tile, 4 K-block fragments
  int t = threadIdx.x, lane = t & 31, wv = t >> 5;
  int row0 = (blockIdx.x * 4 + wv) * 16;
  if (row0 >= Nn) return;  // whole-wave exit; no __syncthreads in this kernel
  int pbase = parity ? 32 : 0, abase = parity ? 0 : 32;

  for (int i = lane; i < 16 * 32; i += 32) {
    int m = i >> 5, c = i & 31;
    Pf[wv][frag_idx(c, m)] = (__bf16)x[(row0 + m) * 64 + pbase + c];
  }
  v16bf ap = *(const v16bf*)&Pf[wv][lane * 16];

  int jrow = (lane >> 4) << 3;  // C-fragment row base for this lane
#pragma unroll
  for (int nt = 0; nt < 8; ++nt) {
    v8f acc = {};
    acc = wmma_bf16(ap, *(const v16bf*)(w1sw + nt * 512 + lane * 16), acc);
    int col = nt * 16 + (lane & 15);
    float bb = b1[col];
#pragma unroll
    for (int j = 0; j < 8; ++j)
      Hf[wv][frag_idx(col, jrow + j)] = (__bf16)fmaxf(acc[j] + bb, 0.f);
  }

  v16bf ah[4];
#pragma unroll
  for (int kb = 0; kb < 4; ++kb) ah[kb] = *(const v16bf*)&Hf[wv][kb * 512 + lane * 16];

  v8f outS[2], outT[2];
#pragma unroll
  for (int nt = 0; nt < 4; ++nt) {
    v8f acc = {};
#pragma unroll
    for (int kb = 0; kb < 4; ++kb)
      acc = wmma_bf16(ah[kb], *(const v16bf*)(w2sw + (nt * 4 + kb) * 512 + lane * 16), acc);
    if (nt < 2) outS[nt] = acc; else outT[nt - 2] = acc;
  }

#pragma unroll
  for (int p = 0; p < 2; ++p) {
    int colS = p * 16 + (lane & 15);  // 0..31 within half
    float bs = b2[colS], bt = b2[32 + colS];
#pragma unroll
    for (int j = 0; j < 8; ++j) {
      int gr = row0 + jrow + j;
      float sv = tanhf(outS[p][j] + bs);
      float tv = outT[p][j] + bt;
      float* px = &x[gr * 64 + abase + colS];
      float xv = *px;
      *px = inverse ? (xv - tv) * __expf(-sv) : xv * __expf(sv) + tv;
    }
  }
}

// ---------------------------------------------------------------------------
// Pool: score = h @ p / ||p||  ; exact-to-22-bits radix top-K selection.
// ---------------------------------------------------------------------------
__global__ void k_pnorm(const float* __restrict__ p, float* __restrict__ pnorm) {
  int lane = threadIdx.x;
  float v0 = p[lane], v1 = p[32 + lane];
  float s = v0 * v0 + v1 * v1;
#pragma unroll
  for (int m = 16; m; m >>= 1) s += __shfl_xor(s, m, 32);
  if (lane == 0) *pnorm = sqrtf(s);
}

__global__ void k_score(const float* __restrict__ h, const float* __restrict__ p,
                        const float* __restrict__ pnorm, float* __restrict__ score,
                        unsigned* __restrict__ key, int Nn) {
  int t = threadIdx.x, lane = t & 31, wid = t >> 5;
  int row = blockIdx.x * 8 + wid;
  if (row >= Nn) return;
  float s = h[row * 64 + lane] * p[lane] + h[row * 64 + 32 + lane] * p[32 + lane];
#pragma unroll
  for (int m = 16; m; m >>= 1) s += __shfl_xor(s, m, 32);
  if (lane == 0) {
    float sc = s / *pnorm;
    score[row] = sc;
    unsigned u = __float_as_uint(sc);
    key[row] = (u & 0x80000000u) ? ~u : (u | 0x80000000u);  // order-preserving
  }
}

__global__ void k_hist1(const unsigned* __restrict__ key, int* __restrict__ hist, int Nn) {
  int n = blockIdx.x * 256 + threadIdx.x;
  if (n < Nn) atomicAdd(&hist[key[n] >> 21], 1);
}
__global__ void k_scan1(const int* __restrict__ hist, int* __restrict__ sel, int Kk) {
  if (threadIdx.x) return;
  int cum = 0;
  for (int b = 2047; b >= 0; --b) {
    int c = hist[b];
    if (cum + c >= Kk) { sel[0] = b; sel[1] = cum; return; }
    cum += c;
  }
  sel[0] = 0; sel[1] = cum;
}
__global__ void k_hist2(const unsigned* __restrict__ key, const int* __restrict__ sel,
                        int* __restrict__ hist2, int Nn) {
  int n = blockIdx.x * 256 + threadIdx.x;
  if (n < Nn && (int)(key[n] >> 21) == sel[0]) atomicAdd(&hist2[(key[n] >> 10) & 0x7FF], 1);
}
__global__ void k_scan2(const int* __restrict__ hist2, int* __restrict__ sel, int Kk) {
  if (threadIdx.x) return;
  int cum = sel[1];
  for (int b = 2047; b >= 0; --b) {
    int c = hist2[b];
    if (cum + c >= Kk) { sel[2] = b; sel[3] = cum; sel[4] = Kk - cum; return; }
    cum += c;
  }
  sel[2] = 0; sel[3] = cum; sel[4] = Kk - cum;
}
__global__ void k_newidx_init(int* __restrict__ newidx, int Nn) {
  int n = blockIdx.x * 256 + threadIdx.x;
  if (n < Nn) newidx[n] = -1;
}
__global__ void k_select(const unsigned* __restrict__ key, const int* __restrict__ sel,
                         int* __restrict__ perm, int* __restrict__ ctrs, int Nn) {
  int n = blockIdx.x * 256 + threadIdx.x;
  if (n >= Nn) return;
  unsigned u = key[n];
  int t1 = (int)(u >> 21), t2 = (int)((u >> 10) & 0x7FF);
  int b1 = sel[0], b2 = sel[2], above2 = sel[3], rem = sel[4];
  bool definite = (t1 > b1) || (t1 == b1 && t2 > b2);
  bool tie = (t1 == b1) && (t2 == b2);
  if (definite) {
    perm[atomicAdd(&ctrs[0], 1)] = n;
  } else if (tie) {
    int p = atomicAdd(&ctrs[1], 1);
    if (p < rem) perm[above2 + p] = n;
  }
}
__global__ void k_gather(const float* __restrict__ h, const float* __restrict__ score,
                         const int* __restrict__ perm, float* __restrict__ xa,
                         int* __restrict__ newidx, int Kk) {
  int idx = blockIdx.x * 256 + threadIdx.x;
  int i = idx >> 6, f = idx & 63;
  if (i >= Kk) return;
  int n = perm[i];
  if (f == 0) newidx[n] = i;
  xa[i * 64 + f] = h[n * 64 + f] * tanhf(score[n]);
}

// ---------------------------------------------------------------------------
// R-GCN layer 2 edge passes (all relations collapse to 0; invalid edges drop)
// ---------------------------------------------------------------------------
__global__ void k_edge_cnt2(const int* __restrict__ src, const int* __restrict__ dst,
                            const int* __restrict__ newidx, float* __restrict__ cnt2, int E) {
  int e = blockIdx.x * 256 + threadIdx.x;
  if (e >= E) return;
  int s2 = newidx[src[e]], d2 = newidx[dst[e]];
  if (s2 >= 0 && d2 >= 0) atomAddF(&cnt2[d2], 1.f);
}
__global__ void k_scatter2(const int* __restrict__ src, const int* __restrict__ dst,
                           const float* __restrict__ conf, const int* __restrict__ newidx,
                           const float* __restrict__ xw2, const float* __restrict__ cnt2,
                           float* __restrict__ h2, int E) {
  int idx = blockIdx.x * 256 + threadIdx.x;
  int e = idx >> 6, f = idx & 63;
  if (e >= E) return;
  int s2 = newidx[src[e]], d2 = newidx[dst[e]];
  if (s2 < 0 || d2 < 0) return;
  float w = conf[e] / fmaxf(cnt2[d2], 1.f);
  atomAddF(&h2[d2 * 64 + f], xw2[s2 * 2 * 64 + f] * w);  // matrix 0 (relation 0)
}
__global__ void k_unpool(const float* __restrict__ xa2, const int* __restrict__ perm,
                         float* __restrict__ out, int Kk) {
  int idx = blockIdx.x * 256 + threadIdx.x;
  int i = idx >> 6, f = idx & 63;
  if (i < Kk) out[perm[i] * 64 + f] = xa2[i * 64 + f];
}

// ---------------------------------------------------------------------------
extern "C" void kernel_launch(void* const* d_in, const int* in_sizes, int n_in,
                              void* d_out, int out_size, void* d_ws, size_t ws_size,
                              hipStream_t stream) {
  const float* x     = (const float*)d_in[0];
  const int*   src   = (const int*)d_in[1];
  const int*   dst   = src + E_;
  const int*   et    = (const int*)d_in[2];
  const float* conf  = (const float*)d_in[3];
  const float* bases1 = (const float*)d_in[4];
  const float* comp1  = (const float*)d_in[5];
  const float* root1  = (const float*)d_in[6];
  const float* bias1  = (const float*)d_in[7];
  const float* bases2 = (const float*)d_in[8];
  const float* comp2  = (const float*)d_in[9];
  const float* root2  = (const float*)d_in[10];
  const float* bias2  = (const float*)d_in[11];
  // coup_f: idx 12 + 4*l {w1,b1,w2,b2} ; coup_i: idx 24 + 4*l
  const float* g1  = (const float*)d_in[36];
  const float* bn1 = (const float*)d_in[37];
  const float* g2  = (const float*)d_in[38];
  const float* bn2 = (const float*)d_in[39];
  const float* poolp = (const float*)d_in[40];
  float* out = (float*)d_out;

  // ---- workspace bump allocator ----
  char* wsp = (char*)d_ws;
  auto alloc = [&](size_t bytes) {
    char* p = wsp;
    wsp += (bytes + 255) & ~size_t(255);
    return p;
  };
  __bf16* Wsw1  = (__bf16*)alloc(17 * F_ * F_ * 2);
  __bf16* Wsw2  = (__bf16*)alloc(2 * F_ * F_ * 2);
  __bf16* cw1sw = (__bf16*)alloc(6 * 32 * H_ * 2);   // 6 layers (3 fwd + 3 inv) w1
  __bf16* cw2sw = (__bf16*)alloc(6 * H_ * F_ * 2);   // 6 layers w2
  float* xw1   = (float*)alloc((size_t)N_ * 17 * 64 * 4);
  float* h     = (float*)alloc((size_t)N_ * 64 * 4);
  float* cnt1  = (float*)alloc((size_t)N_ * R_ * 4);
  float* score = (float*)alloc(N_ * 4);
  unsigned* key = (unsigned*)alloc(N_ * 4);
  int* perm    = (int*)alloc(K_ * 4);
  int* newidx  = (int*)alloc(N_ * 4);
  int* hist    = (int*)alloc(2048 * 4);
  int* hist2   = (int*)alloc(2048 * 4);
  int* sel     = (int*)alloc(8 * 4);
  int* ctrs    = (int*)alloc(2 * 4);
  float* pnorm = (float*)alloc(4);
  float* xa    = (float*)alloc((size_t)KPAD_ * 64 * 4);      // padded rows
  float* xw2   = (float*)alloc((size_t)KPAD_ * 2 * 64 * 4);  // padded rows
  float* cnt2  = (float*)alloc(K_ * 4);
  float* h2    = (float*)alloc((size_t)K_ * 64 * 4);

  const int nt1 = N_ / 16;        // 3125 (N divisible by 16)
  const int nt2 = KPAD_ / 16;     // 1563
  const int ebk = (E_ + 255) / 256;
  const int nbk = (N_ + 255) / 256;

  // ---- parameter prep (basis combination + fragment-major bf16 swizzle) ----
  k_prep_W_swz<<<17, 256, 0, stream>>>(bases1, comp1, root1, Wsw1, 16);
  k_prep_W_swz<<<2, 256, 0, stream>>>(bases2, comp2, root2, Wsw2, 1);
  for (int l = 0; l < 3; ++l) {
    k_f2bf_swz<<<16, 256, 0, stream>>>((const float*)d_in[12 + 4 * l + 0], cw1sw + l * 4096, 32, 128);
    k_f2bf_swz<<<32, 256, 0, stream>>>((const float*)d_in[12 + 4 * l + 2], cw2sw + l * 8192, 128, 64);
    k_f2bf_swz<<<16, 256, 0, stream>>>((const float*)d_in[24 + 4 * l + 0], cw1sw + (3 + l) * 4096, 32, 128);
    k_f2bf_swz<<<32, 256, 0, stream>>>((const float*)d_in[24 + 4 * l + 2], cw2sw + (3 + l) * 8192, 128, 64);
  }

  // ---- R-GCN layer 1: xw = x @ {W_0..W_15, root}; mean scatter; LN; ReLU ----
  k_xw<<<nt1, 256, 0, stream>>>(x, Wsw1, xw1, 17);
  k_zero_f<<<(N_ * R_ + 255) / 256, 256, 0, stream>>>(cnt1, N_ * R_);
  k_edge_cnt1<<<ebk, 256, 0, stream>>>(src, dst, et, cnt1, E_);
  k_init_h<<<(N_ * 64 + 255) / 256, 256, 0, stream>>>(xw1, bias1, h, N_, 17);
  k_scatter1<<<(E_ * 64) / 256, 256, 0, stream>>>(src, dst, et, conf, xw1, cnt1, h, E_);
  k_ln_relu<<<(N_ + 7) / 8, 256, 0, stream>>>(h, g1, bn1, h, N_);

  // ---- forward coupling (3 layers, in place on h) ----
  for (int l = 0; l < 3; ++l)
    k_coupling<<<(nt1 + 3) / 4, 128, 0, stream>>>(h, cw1sw + l * 4096,
        (const float*)d_in[12 + 4 * l + 1], cw2sw + l * 8192,
        (const float*)d_in[12 + 4 * l + 3], l & 1, 0, N_);

  // ---- top-K pooling (order-free set selection; output is perm-invariant) ----
  k_pnorm<<<1, 32, 0, stream>>>(poolp, pnorm);
  k_score<<<(N_ + 7) / 8, 256, 0, stream>>>(h, poolp, pnorm, score, key, N_);
  k_zero_i<<<8, 256, 0, stream>>>(hist, 2048);
  k_zero_i<<<8, 256, 0, stream>>>(hist2, 2048);
  k_zero_i<<<1, 256, 0, stream>>>(ctrs, 2);
  k_hist1<<<nbk, 256, 0, stream>>>(key, hist, N_);
  k_scan1<<<1, 32, 0, stream>>>(hist, sel, K_);
  k_hist2<<<nbk, 256, 0, stream>>>(key, sel, hist2, N_);
  k_scan2<<<1, 32, 0, stream>>>(hist2, sel, K_);
  k_newidx_init<<<nbk, 256, 0, stream>>>(newidx, N_);
  k_select<<<nbk, 256, 0, stream>>>(key, sel, perm, ctrs, N_);
  // zero the 8 pad rows of xa, then gather real rows
  k_zero_f<<<((KPAD_ - K_) * 64 + 255) / 256, 256, 0, stream>>>(xa + K_ * 64, (KPAD_ - K_) * 64);
  k_gather<<<(K_ * 64 + 255) / 256, 256, 0, stream>>>(h, score, perm, xa, newidx, K_);

  // ---- R-GCN layer 2 (relation 0 only) + LN + ReLU ----
  k_xw<<<nt2, 256, 0, stream>>>(xa, Wsw2, xw2, 2);
  k_zero_f<<<(K_ + 255) / 256, 256, 0, stream>>>(cnt2, K_);
  k_edge_cnt2<<<ebk, 256, 0, stream>>>(src, dst, newidx, cnt2, E_);
  k_init_h<<<(K_ * 64 + 255) / 256, 256, 0, stream>>>(xw2, bias2, h2, K_, 2);
  k_scatter2<<<(E_ * 64) / 256, 256, 0, stream>>>(src, dst, conf, newidx, xw2, cnt2, h2, E_);
  k_ln_relu<<<(K_ + 7) / 8, 256, 0, stream>>>(h2, g2, bn2, h2, K_);

  // ---- unpool (zero fill) + inverse coupling into d_out ----
  k_zero_f<<<(N_ * 64 + 255) / 256, 256, 0, stream>>>(out, N_ * 64);
  k_unpool<<<(K_ * 64 + 255) / 256, 256, 0, stream>>>(h2, perm, out, K_);
  for (int l = 2; l >= 0; --l)
    k_coupling<<<(nt1 + 3) / 4, 128, 0, stream>>>(out, cw1sw + (3 + l) * 4096,
        (const float*)d_in[24 + 4 * l + 1], cw2sw + (3 + l) * 8192,
        (const float*)d_in[24 + 4 * l + 3], l & 1, 1, N_);

  (void)in_sizes; (void)n_in; (void)out_size; (void)ws_size;
}
